// DCCcnn_89687507075072
// MI455X (gfx1250) — matrix operand order repaired and checked by reading, not compiled, not run on hardware
//
#include <hip/hip_runtime.h>
#include <hip/hip_bf16.h>

typedef __attribute__((ext_vector_type(16))) _Float16 v16h;
typedef __attribute__((ext_vector_type(8)))  _Float16 v8h;
typedef __attribute__((ext_vector_type(8)))  float    v8f;

#define NB   4      // batch
#define CIN  1024
#define CMID 256
#define HW   4096   // 64*64
#define CF   768    // concat channels

// Toggle for the gfx1250 async global->LDS staging path.
#define ASYNC_STAGE 1

// ---------------------------------------------------------------------------
// gfx1250 async global->LDS b128 copy (ASYNCcnt-tracked DMA, no VGPR bounce).
// LDS byte offset = low 32 bits of the generic pointer to __shared__.
// ---------------------------------------------------------------------------
__device__ __forceinline__ void async_b128(const _Float16* g, _Float16* l) {
#if ASYNC_STAGE
    unsigned lds_off = (unsigned)(unsigned long long)l;
    asm volatile("global_load_async_to_lds_b128 %0, %1, off"
                 :: "v"(lds_off), "v"(g) : "memory");
#else
    *(v8h*)l = *(const v8h*)g;
#endif
}
__device__ __forceinline__ void wait_async0() {
#if ASYNC_STAGE
    asm volatile("s_wait_asynccnt 0x0" ::: "memory");
#endif
}

// ---------------------------------------------------------------------------
// flat f32 -> f16 convert
// ---------------------------------------------------------------------------
__global__ __launch_bounds__(256) void f32_to_f16(const float* __restrict__ in,
                                                  _Float16* __restrict__ out, int n) {
    int i = blockIdx.x * 256 + threadIdx.x;
    if (i < n) out[i] = (_Float16)in[i];
}

// ---------------------------------------------------------------------------
// Generic WMMA GEMM:  C = alpha * (A @ B) + beta * Cin + bias  (+ f16 mirror)
//   A: (M x K) f16 row-major.
//   B: BMODE==0 ("KN"): B[k][n] = Bm[k*ldb + n]
//      BMODE==1 ("NK"): B[k][n] = Bm[n*ldb + k]  (C = A @ Bm^T)
//   Block tile 64(M) x 128(N), K-stage 64 (2 WMMA K-steps, 8 wmma per stage).
//   8 waves: 2(M) x 4(N), each wave 32x32 macro-tile.
//   Requires M%64==0, N%128==0, K%64==0 (true for all GEMMs here).
// ---------------------------------------------------------------------------
template<int BMODE, bool HAS_C16, bool HAS_CIN, bool HAS_BIAS, bool APTR>
__global__ __launch_bounds__(256) void gemm_wmma_f16(
    const _Float16* __restrict__ A,  int lda,  long long sA,
    const _Float16* __restrict__ Bm, int ldb,  long long sB,
    float*          __restrict__ C,  int ldc,  long long sC,
    _Float16*       __restrict__ C16,int ldc16,long long sC16,
    const float*    __restrict__ Cin,int ldcin,long long sCin,
    const float*    __restrict__ bias,
    const float*    __restrict__ alpha_ptr, float alpha_imm, float beta,
    int Kdim)
{
    __shared__ __align__(16) _Float16 Al[64 * 72];   // 64 x 64 halves, row stride 72
    __shared__ __align__(16) _Float16 Bl[128 * 72];  // transposed [n][k], stride 72

    const int tid   = threadIdx.x;
    const int batch = blockIdx.z;
    const int m0    = blockIdx.y * 64;
    const int n0    = blockIdx.x * 128;

    const int lane = tid & 31;
    const int wid  = tid >> 5;
    const int mw   = (wid & 1) * 32;     // wave M offset in block tile
    const int nw   = (wid >> 1) * 32;    // wave N offset in block tile
    const int l15  = lane & 15;
    const int hi   = lane >> 4;

    v8f zero8 = {0.f, 0.f, 0.f, 0.f, 0.f, 0.f, 0.f, 0.f};
    v8f acc[2][2];
    acc[0][0] = zero8; acc[0][1] = zero8; acc[1][0] = zero8; acc[1][1] = zero8;

    const _Float16* Ab = A  + sA * batch;
    const _Float16* Bb = Bm + sB * batch;

    // staging coordinates (uniform per thread across iterations)
    const int arow = tid >> 2, acb = (tid & 3) * 16;          // A: 2 x b128
    const int bk   = tid >> 3, bnb = (tid & 7) * 16;          // B KN transpose
    const int bn   = tid >> 1, bkb = (tid & 1) * 32;          // B NK copy: 4 x b128

    for (int k0 = 0; k0 < Kdim; k0 += 64) {
        __syncthreads();
        // ---- stage A tile (64 x 64 halves): straight copy -> async DMA
        {
            const _Float16* src = Ab + (size_t)(m0 + arow) * lda + k0 + acb;
            _Float16*       dst = Al + arow * 72 + acb;
            async_b128(src,     dst);
            async_b128(src + 8, dst + 8);
        }
        // ---- stage B tile (64 x 128)
        if (BMODE == 0) {
            // needs transpose: VALU scatter into Bl[n][k]
#pragma unroll
            for (int kh = 0; kh < 2; ++kh) {
                const int krow = bk + kh * 32;
                const _Float16* src = Bb + (size_t)(k0 + krow) * ldb + n0 + bnb;
                v8h b0 = *(const v8h*)(src);
                v8h b1 = *(const v8h*)(src + 8);
#pragma unroll
                for (int j = 0; j < 8; ++j) {
                    Bl[(bnb + j)     * 72 + krow] = b0[j];
                    Bl[(bnb + 8 + j) * 72 + krow] = b1[j];
                }
            }
        } else {
            // straight copy -> async DMA
            const _Float16* src = Bb + (size_t)(n0 + bn) * ldb + k0 + bkb;
            _Float16*       dst = Bl + bn * 72 + bkb;
#pragma unroll
            for (int s = 0; s < 4; ++s) async_b128(src + 8 * s, dst + 8 * s);
        }
        wait_async0();
        __syncthreads();

        // ---- two WMMA K-steps out of the 64-wide stage
#pragma unroll
        for (int kk = 0; kk < 64; kk += 32) {
            v16h afr[2], bfr[2];
#pragma unroll
            for (int i = 0; i < 2; ++i) {
                // A 16x32: lane<16 holds K 0-7/16-23, lane>=16 holds K 8-15/24-31
                const _Float16* ar = Al + (mw + i * 16 + l15) * 72 + kk;
                v8h a0 = *(const v8h*)(ar + hi * 8);
                v8h a1 = *(const v8h*)(ar + 16 + hi * 8);
#pragma unroll
                for (int e = 0; e < 8; ++e) { afr[i][e] = a0[e]; afr[i][8 + e] = a1[e]; }
                // B 32x16: lanes 0-15 hold K 0-7/16-23 (N=lane), lanes 16-31 K 8-15/24-31
                const _Float16* br = Bl + (nw + i * 16 + l15) * 72 + kk;
                v8h b0 = *(const v8h*)(br + hi * 8);
                v8h b1 = *(const v8h*)(br + 16 + hi * 8);
#pragma unroll
                for (int e = 0; e < 8; ++e) { bfr[i][e] = b0[e]; bfr[i][8 + e] = b1[e]; }
            }
#pragma unroll
            for (int i = 0; i < 2; ++i)
#pragma unroll
                for (int j = 0; j < 2; ++j)
                    acc[i][j] = __builtin_amdgcn_wmma_f32_16x16x32_f16(
                        false, afr[i], false, bfr[j], (short)0, acc[i][j], false, false);
        }
    }

    // ---- epilogue (fully compile-time configured, no runtime branches)
    // C/D layout: VGPR r -> lanes 0-15 M=r, lanes 16-31 M=8+r; N = lane&15
    const float alpha = APTR ? *alpha_ptr : alpha_imm;
    const int hi8 = hi * 8;
#pragma unroll
    for (int i = 0; i < 2; ++i) {
#pragma unroll
        for (int j = 0; j < 2; ++j) {
            const int mb = m0 + mw + i * 16 + hi8;
            const int nn = n0 + nw + j * 16 + l15;
#pragma unroll
            for (int r = 0; r < 8; ++r) {
                const int m = mb + r;
                float v = alpha * acc[i][j][r];
                if (HAS_CIN)  v += beta * Cin[sCin * batch + (size_t)m * ldcin + nn];
                if (HAS_BIAS) v += bias[m];
                C[sC * batch + (size_t)m * ldc + nn] = v;
                if (HAS_C16)  C16[sC16 * batch + (size_t)m * ldc16 + nn] = (_Float16)v;
            }
        }
    }
}

// ---------------------------------------------------------------------------
// row softmax (f32 in, f16 out); one block per row
// ---------------------------------------------------------------------------
__global__ __launch_bounds__(256) void softmax_rows(const float* __restrict__ in,
                                                    _Float16* __restrict__ out, int cols) {
    __shared__ float red[256];
    const float* row = in + (size_t)blockIdx.x * cols;
    _Float16* orow   = out + (size_t)blockIdx.x * cols;
    int tid = threadIdx.x;
    float mx = -1e30f;
    for (int c = tid; c < cols; c += 256) mx = fmaxf(mx, row[c]);
    red[tid] = mx; __syncthreads();
    for (int s = 128; s > 0; s >>= 1) { if (tid < s) red[tid] = fmaxf(red[tid], red[tid + s]); __syncthreads(); }
    mx = red[0]; __syncthreads();
    float sum = 0.f;
    for (int c = tid; c < cols; c += 256) sum += __expf(row[c] - mx);
    red[tid] = sum; __syncthreads();
    for (int s = 128; s > 0; s >>= 1) { if (tid < s) red[tid] += red[tid + s]; __syncthreads(); }
    float inv = 1.f / red[0];
    for (int c = tid; c < cols; c += 256) orow[c] = (_Float16)(__expf(row[c] - mx) * inv);
}

// ---------------------------------------------------------------------------
// dilated 3x3 conv: s[dil][n][co=0..8][pix] from xc (n,256,64,64)
// weights LDS-staged in ci-chunks of 32
// ---------------------------------------------------------------------------
__constant__ int kDils[4] = {1, 3, 6, 12};

__global__ __launch_bounds__(256) void adc_conv(const float* __restrict__ xc,
                                                const float* __restrict__ w,
                                                float* __restrict__ s) {
    const int dil = blockIdx.z, n = blockIdx.y;
    const int d = kDils[dil];
    const int tid = threadIdx.x;
    const int pix = blockIdx.x * 256 + tid;
    const int y = pix >> 6, x = pix & 63;
    __shared__ float wl[32 * 81]; // [co][ci][tap] = co*288 + ci*9 + tap
    float acc[9] = {0,0,0,0,0,0,0,0,0};
    const float* xcb = xc + (size_t)n * CMID * HW;
    for (int c0 = 0; c0 < CMID; c0 += 32) {
        __syncthreads();
        for (int idx = tid; idx < 32 * 81; idx += 256) {
            int co = idx / 288, r = idx % 288;
            wl[idx] = w[(size_t)dil * 20736 + (size_t)co * 2304 + (size_t)c0 * 9 + r];
        }
        __syncthreads();
        for (int ci = 0; ci < 32; ++ci) {
            const float* xp = xcb + (size_t)(c0 + ci) * HW;
            float p[9];
#pragma unroll
            for (int ky = 0; ky < 3; ++ky)
#pragma unroll
                for (int kx = 0; kx < 3; ++kx) {
                    int yy = y + (ky - 1) * d, xx = x + (kx - 1) * d;
                    p[ky * 3 + kx] = (yy >= 0 && yy < 64 && xx >= 0 && xx < 64)
                                       ? xp[yy * 64 + xx] : 0.f;
                }
#pragma unroll
            for (int co = 0; co < 9; ++co) {
                const float* ww = wl + co * 288 + ci * 9;
#pragma unroll
                for (int t = 0; t < 9; ++t) acc[co] += ww[t] * p[t];
            }
        }
    }
    float* sp = s + ((size_t)(dil * NB + n) * 9) * HW + pix;
#pragma unroll
    for (int co = 0; co < 9; ++co) sp[(size_t)co * HW] = acc[co];
}

// ---------------------------------------------------------------------------
// BN training stats per (dil, co): mean & rsqrt(var+eps) over n,h,w
// ---------------------------------------------------------------------------
__global__ __launch_bounds__(256) void bn_stats(const float* __restrict__ s,
                                                float* __restrict__ meanv,
                                                float* __restrict__ invstd) {
    const int b = blockIdx.x;            // dil*9 + co
    const int dil = b / 9, co = b % 9;
    __shared__ float rs[256], rq[256];
    const int tid = threadIdx.x;
    float sum = 0.f, sq = 0.f;
    for (int n = 0; n < NB; ++n) {
        const float* p = s + ((size_t)(dil * NB + n) * 9 + co) * HW;
        for (int i = tid; i < HW; i += 256) { float v = p[i]; sum += v; sq += v * v; }
    }
    rs[tid] = sum; rq[tid] = sq; __syncthreads();
    for (int st = 128; st > 0; st >>= 1) {
        if (tid < st) { rs[tid] += rs[tid + st]; rq[tid] += rq[tid + st]; }
        __syncthreads();
    }
    if (tid == 0) {
        float m = rs[0] * (1.f / 16384.f);
        float var = rq[0] * (1.f / 16384.f) - m * m;
        meanv[b] = m;
        invstd[b] = rsqrtf(var + 1e-5f);
    }
}

// ---------------------------------------------------------------------------
// BN + softmax over 9 taps + unfold-weighted sum -> feat channels [dil*64, +64)
// ---------------------------------------------------------------------------
__global__ __launch_bounds__(256) void adc_unfold(const float* __restrict__ s,
                                                  const float* __restrict__ meanv,
                                                  const float* __restrict__ invstd,
                                                  const float* __restrict__ bnw,
                                                  const float* __restrict__ bnb,
                                                  const float* __restrict__ xc,
                                                  _Float16* __restrict__ feat) {
    const int dil = blockIdx.z, n = blockIdx.y;
    const int d = kDils[dil];
    const int tid = threadIdx.x;
    const int pix = blockIdx.x * 256 + tid;
    const int y = pix >> 6, x = pix & 63;

    float sig[9]; float mx = -1e30f;
    const float* sp = s + ((size_t)(dil * NB + n) * 9) * HW + pix;
#pragma unroll
    for (int t = 0; t < 9; ++t) {
        float v = sp[(size_t)t * HW];
        v = (v - meanv[dil * 9 + t]) * invstd[dil * 9 + t] * bnw[dil * 9 + t] + bnb[dil * 9 + t];
        sig[t] = v; mx = fmaxf(mx, v);
    }
    float sum = 0.f;
#pragma unroll
    for (int t = 0; t < 9; ++t) { sig[t] = __expf(sig[t] - mx); sum += sig[t]; }
    float inv = 1.f / sum;
#pragma unroll
    for (int t = 0; t < 9; ++t) sig[t] *= inv;

    int offs[9]; bool val[9];
#pragma unroll
    for (int ky = 0; ky < 3; ++ky)
#pragma unroll
        for (int kx = 0; kx < 3; ++kx) {
            int yy = y + (ky - 1) * d, xx = x + (kx - 1) * d;
            val[ky * 3 + kx]  = (yy >= 0 && yy < 64 && xx >= 0 && xx < 64);
            offs[ky * 3 + kx] = yy * 64 + xx;
        }
    const float* xcb = xc + ((size_t)n * CMID + dil * 64) * HW;
    _Float16* fb = feat + ((size_t)n * CF + dil * 64) * HW + pix;
    for (int c = 0; c < 64; ++c) {
        const float* xp = xcb + (size_t)c * HW;
        float a = 0.f;
#pragma unroll
        for (int t = 0; t < 9; ++t) a += val[t] ? xp[offs[t]] * sig[t] : 0.f;
        fb[(size_t)c * HW] = (_Float16)a;
    }
}

// ---------------------------------------------------------------------------
// per-(n,c) spatial mean of x
// ---------------------------------------------------------------------------
__global__ __launch_bounds__(256) void pool_mean(const float* __restrict__ x,
                                                 float* __restrict__ gm) {
    const int b = blockIdx.x;            // n*1024 + c
    __shared__ float rs[256];
    const float* p = x + (size_t)b * HW;
    const int tid = threadIdx.x;
    float sum = 0.f;
    for (int i = tid; i < HW; i += 256) sum += p[i];
    rs[tid] = sum; __syncthreads();
    for (int st = 128; st > 0; st >>= 1) { if (tid < st) rs[tid] += rs[tid + st]; __syncthreads(); }
    if (tid == 0) gm[b] = rs[0] * (1.f / (float)HW);
}

// ---------------------------------------------------------------------------
// x4[n][co] = reduce_w[co] . gm[n] + reduce_b[co]; broadcast-fill feat ch 256+co
// ---------------------------------------------------------------------------
__global__ __launch_bounds__(256) void reduce_fill(const float* __restrict__ gm,
                                                   const float* __restrict__ rw,
                                                   const float* __restrict__ rb,
                                                   _Float16* __restrict__ feat) {
    const int b = blockIdx.x;            // n*256 + co
    const int n = b >> 8, co = b & 255;
    __shared__ float rs[256];
    const int tid = threadIdx.x;
    const float* g = gm + (size_t)n * CIN;
    const float* w = rw + (size_t)co * CIN;
    float sum = 0.f;
    for (int i = tid; i < CIN; i += 256) sum += g[i] * w[i];
    rs[tid] = sum; __syncthreads();
    for (int st = 128; st > 0; st >>= 1) { if (tid < st) rs[tid] += rs[tid + st]; __syncthreads(); }
    _Float16 hv = (_Float16)(rs[0] + rb[co]);
    _Float16* f = feat + ((size_t)n * CF + 256 + co) * HW;
    for (int i = tid; i < HW; i += 256) f[i] = hv;
}

// ---------------------------------------------------------------------------
extern "C" void kernel_launch(void* const* d_in, const int* in_sizes, int n_in,
                              void* d_out, int out_size, void* d_ws, size_t ws_size,
                              hipStream_t stream) {
    const float* x     = (const float*)d_in[0];
    const float* cam_w = (const float*)d_in[1];
    const float* cam_b = (const float*)d_in[2];
    const float* gamma = (const float*)d_in[3];
    const float* adc_w = (const float*)d_in[4];
    const float* bn_w  = (const float*)d_in[5];
    const float* bn_b  = (const float*)d_in[6];
    const float* red_w = (const float*)d_in[7];
    const float* red_b = (const float*)d_in[8];
    const float* out_w = (const float*)d_in[9];
    const float* out_b = (const float*)d_in[10];
    float* out = (float*)d_out;

    char* ws = (char*)d_ws;
    size_t off = 0;
    auto take = [&](size_t bytes) -> char* {
        off = (off + 255) & ~(size_t)255;
        char* p = ws + off; off += bytes; return p;
    };
    _Float16* x16    = (_Float16*)take((size_t)NB * CIN * HW * 2);   // 32 MB
    _Float16* camw16 = (_Float16*)take((size_t)CMID * CIN * 2);
    _Float16* outw16 = (_Float16*)take((size_t)CMID * CF * 2);
    float*    q      = (float*)   take((size_t)NB * CMID * HW * 4);  // q, then xc in-place
    _Float16* q16    = (_Float16*)take((size_t)NB * CMID * HW * 2);
    float*    energy = (float*)   take((size_t)NB * CMID * CIN * 4);
    _Float16* attn16 = (_Float16*)take((size_t)NB * CMID * CIN * 2);
    float*    sbuf   = (float*)   take((size_t)4 * NB * 9 * HW * 4);
    float*    meanv  = (float*)   take(36 * 4);
    float*    invstd = (float*)   take(36 * 4);
    float*    gm     = (float*)   take((size_t)NB * CIN * 4);
    _Float16* feat   = (_Float16*)take((size_t)NB * CF * HW * 2);    // 24 MB
    (void)ws_size; (void)in_sizes; (void)n_in; (void)out_size;

    // 1) f16 conversions
    f32_to_f16<<<(NB * CIN * HW + 255) / 256, 256, 0, stream>>>(x, x16, NB * CIN * HW);
    f32_to_f16<<<(CMID * CIN + 255) / 256, 256, 0, stream>>>(cam_w, camw16, CMID * CIN);
    f32_to_f16<<<(CMID * CF + 255) / 256, 256, 0, stream>>>(out_w, outw16, CMID * CF);

    // 2) q = cam_w @ x + cam_b   (M=256,K=1024,N=4096)  -> q (f32) and q16
    gemm_wmma_f16<0, true, false, true, false>
        <<<dim3(HW / 128, CMID / 64, NB), 256, 0, stream>>>(
        camw16, CIN, 0LL,
        x16, HW, (long long)CIN * HW,
        q, HW, (long long)CMID * HW,
        q16, HW, (long long)CMID * HW,
        nullptr, 0, 0LL,
        cam_b, nullptr, 1.0f, 0.0f, CIN);

    // 3) energy = (q @ x^T) / 64  (M=256,K=4096,N=1024)
    gemm_wmma_f16<1, false, false, false, false>
        <<<dim3(CIN / 128, CMID / 64, NB), 256, 0, stream>>>(
        q16, HW, (long long)CMID * HW,
        x16, HW, (long long)CIN * HW,
        energy, CIN, (long long)CMID * CIN,
        nullptr, 0, 0LL,
        nullptr, 0, 0LL,
        nullptr, nullptr, 0.015625f, 0.0f, HW);

    // 4) attn = softmax(energy) over last axis (1024), emit f16
    softmax_rows<<<NB * CMID, 256, 0, stream>>>(energy, attn16, CIN);

    // 5) xc = gamma * (attn @ x) + q  (in place over q); f16 copy into feat ch 512..767
    gemm_wmma_f16<0, true, true, false, true>
        <<<dim3(HW / 128, CMID / 64, NB), 256, 0, stream>>>(
        attn16, CIN, (long long)CMID * CIN,
        x16, HW, (long long)CIN * HW,
        q, HW, (long long)CMID * HW,
        feat + (size_t)512 * HW, HW, (long long)CF * HW,
        q, HW, (long long)CMID * HW,
        nullptr, gamma, 1.0f, 1.0f, CIN);

    // 6) four dilated 3x3 convs -> s
    adc_conv<<<dim3(HW / 256, NB, 4), 256, 0, stream>>>(q, adc_w, sbuf);
    // 7) BN training stats
    bn_stats<<<36, 256, 0, stream>>>(sbuf, meanv, invstd);
    // 8) BN + softmax(9) + unfold weighted sum -> feat ch 0..255
    adc_unfold<<<dim3(HW / 256, NB, 4), 256, 0, stream>>>(
        sbuf, meanv, invstd, bn_w, bn_b, q, feat);

    // 9) global context -> feat ch 256..511
    pool_mean<<<NB * CIN, 256, 0, stream>>>(x, gm);
    reduce_fill<<<NB * CMID, 256, 0, stream>>>(gm, red_w, red_b, feat);

    // 10) out = out_w @ feat + out_b  (M=256,K=768,N=4096)
    gemm_wmma_f16<0, false, false, true, false>
        <<<dim3(HW / 128, CMID / 64, NB), 256, 0, stream>>>(
        outw16, CF, 0LL,
        feat, HW, (long long)CF * HW,
        out, HW, (long long)CMID * HW,
        nullptr, 0, 0LL,
        nullptr, 0, 0LL,
        out_b, nullptr, 1.0f, 0.0f, CF);
}
